// SCNN_45457933860952
// MI455X (gfx1250) — compile-verified, matrix operand order checked by reading, not run on hardware
//
#include <hip/hip_runtime.h>

typedef __attribute__((ext_vector_type(16))) __bf16 v16bf;
typedef __attribute__((ext_vector_type(8)))  float  v8f;

#define CCH   128          // channels
#define HW    128          // H and W
#define NB    8            // batch
#define KW9   9            // conv taps
#define PITCH 136          // padded LDS row length (elements) -> 272B, breaks 256B bank period
#define ROWS  136          // 128 + 2*4 zero pad rows
#define PART_BYTES (ROWS * PITCH * 2)   // 36992 B per (buffer, hi/lo) part
#define PLANE 2097152      // 128*128*128 elements per batch

union FragU { v16bf v; uint4 q[2]; };

// ---------------------------------------------------------------------------
// Pack conv weights K[dw][ci][co] (f32) into WMMA B-fragment blobs (bf16 hi/lo).
// Fragment index: ((dw*4 + cg)*8 + ntile), blob = 32 lanes x 16 bf16 (1 KB).
// Lane half 0 holds K-local {0..7,16..23}, half 1 holds {8..15,24..31}
// (mirrors the ISA 16-bit operand layout).
// ---------------------------------------------------------------------------
__global__ void scnn_wprep(const float* __restrict__ K,
                           __bf16* __restrict__ hi, __bf16* __restrict__ lo)
{
    int t = blockIdx.x * blockDim.x + threadIdx.x;
    if (t >= 288 * 32) return;
    int lane = t & 31;
    int frag = t >> 5;
    int nt = frag & 7;
    int cg = (frag >> 3) & 3;
    int dw = frag >> 5;
    int halfk = lane >> 4;
    int co = nt * 16 + (lane & 15);
    __bf16* ph = hi + (size_t)frag * 512 + lane * 16;
    __bf16* pl = lo + (size_t)frag * 512 + lane * 16;
    #pragma unroll
    for (int j = 0; j < 16; ++j) {
        int kl = (j < 8) ? (halfk * 8 + j) : (16 + halfk * 8 + (j - 8));
        int ci = cg * 32 + kl;
        float w = K[((size_t)dw * CCH + ci) * CCH + co];
        __bf16 h = (__bf16)w;
        ph[j] = h;
        pl[j] = (__bf16)(w - (float)h);
    }
}

// ---------------------------------------------------------------------------
// Tiled transposes NCHW <-> NHWC (both sides coalesced through LDS).
// ---------------------------------------------------------------------------
__global__ __launch_bounds__(256) void scnn_t_in(const float* __restrict__ X,
                                                 float* __restrict__ Y)
{
    __shared__ float tile[32][33];
    int bh = blockIdx.x;               // b*128 + h
    int b = bh >> 7, h = bh & 127;
    int tw = blockIdx.y & 3, tc = blockIdx.y >> 2;
    int x0 = threadIdx.x, y0 = threadIdx.y;
    #pragma unroll
    for (int k = 0; k < 4; ++k) {
        int c = tc * 32 + y0 + k * 8;
        int w = tw * 32 + x0;
        tile[y0 + k * 8][x0] = X[(size_t)b * PLANE + (size_t)c * 16384 + h * 128 + w];
    }
    __syncthreads();
    #pragma unroll
    for (int k = 0; k < 4; ++k) {
        int w = tw * 32 + y0 + k * 8;
        int c = tc * 32 + x0;
        Y[(size_t)b * PLANE + (size_t)h * 16384 + w * 128 + c] = tile[x0][y0 + k * 8];
    }
}

__global__ __launch_bounds__(256) void scnn_t_out(const float* __restrict__ Y,
                                                  float* __restrict__ O)
{
    __shared__ float tile[32][33];
    int bh = blockIdx.x;
    int b = bh >> 7, h = bh & 127;
    int tw = blockIdx.y & 3, tc = blockIdx.y >> 2;
    int x0 = threadIdx.x, y0 = threadIdx.y;
    #pragma unroll
    for (int k = 0; k < 4; ++k) {
        int w = tw * 32 + y0 + k * 8;
        int c = tc * 32 + x0;
        tile[y0 + k * 8][x0] = Y[(size_t)b * PLANE + (size_t)h * 16384 + w * 128 + c];
    }
    __syncthreads();
    #pragma unroll
    for (int k = 0; k < 4; ++k) {
        int w = tw * 32 + x0;
        int c = tc * 32 + y0 + k * 8;
        O[(size_t)b * PLANE + (size_t)c * 16384 + h * 128 + w] = tile[x0][y0 + k * 8];
    }
}

// ---------------------------------------------------------------------------
// Directional scan. One workgroup (16 waves) per batch; prev row in LDS
// (bf16 hi/lo, ping-pong). Each wave owns a 2x2 block of 16x16 tiles:
// m tiles {2mg,2mg+1}, n tiles {2ng,2ng+1} with mg = wave>>2, ng = wave&3.
// Per K-chunk: 4 LDS A-frags + 4 VMEM B-frags feed 12 WMMAs (balanced ports).
// Split-bf16: acc += Ahi*Bhi + Alo*Bhi + Ahi*Blo  (f32-grade precision).
// su = element stride along scan axis, sv = stride along conv axis.
// ---------------------------------------------------------------------------
__global__ __launch_bounds__(512)
void scnn_scan(float* __restrict__ Y,
               const __bf16* __restrict__ fragHi,
               const __bf16* __restrict__ fragLo,
               int su, int sv, int rev)
{
    extern __shared__ char smem[];
    const int tid  = threadIdx.x;
    const int lane = tid & 31;
    const int wave = tid >> 5;          // 0..15
    float* Yb = Y + (size_t)blockIdx.x * PLANE;

    // zero both ping-pong buffers (covers the 4-row zero pads)
    unsigned int* sw = (unsigned int*)smem;
    for (int i = tid; i < 4 * PART_BYTES / 4; i += 512) sw[i] = 0u;
    __syncthreads();

    // load first line into buffer 0 (hi/lo)
    const int u0 = rev ? (HW - 1) : 0;
    {
        const float* row = Yb + (size_t)u0 * su;
        __bf16* hi = (__bf16*)(smem);
        __bf16* lo = (__bf16*)(smem + PART_BYTES);
        int base = tid * 32;              // 512*32 == 16384 exactly
        int v = base >> 7, c0 = base & 127;
        #pragma unroll
        for (int j = 0; j < 32; ++j) {
            float x = row[(size_t)v * sv + c0 + j];
            __bf16 h = (__bf16)x;
            hi[(v + 4) * PITCH + c0 + j] = h;
            lo[(v + 4) * PITCH + c0 + j] = (__bf16)(x - (float)h);
        }
    }
    __syncthreads();

    const int mg   = wave >> 2;         // m-pair 0..3
    const int ng   = wave & 3;          // n-pair 0..3
    const int half = lane >> 4;
    const int li   = lane & 15;

    int p = 0;
    for (int step = 1; step < HW; ++step) {
        const int u = rev ? (HW - 1 - step) : step;
        const char* Ah = smem + (size_t)(p * 2) * PART_BYTES;
        const char* Al = Ah + PART_BYTES;
        float* Yrow = Yb + (size_t)u * su;

        // preload residual ('cur') values: latency hides behind the K-loop
        float cur[2][2][8];
        #pragma unroll
        for (int mt = 0; mt < 2; ++mt)
            #pragma unroll
            for (int nt = 0; nt < 2; ++nt) {
                const int co = (ng * 2 + nt) * 16 + li;
                #pragma unroll
                for (int r = 0; r < 8; ++r) {
                    const int w = (mg * 2 + mt) * 16 + half * 8 + r;
                    cur[mt][nt][r] = Yrow[(size_t)w * sv + co];
                }
            }

        v8f acc[2][2];
        #pragma unroll
        for (int mt = 0; mt < 2; ++mt)
            #pragma unroll
            for (int nt = 0; nt < 2; ++nt)
                acc[mt][nt] = (v8f){0.f, 0.f, 0.f, 0.f, 0.f, 0.f, 0.f, 0.f};

        for (int dw = 0; dw < KW9; ++dw) {
            // prefetch next tap's first B line into the caches
            if (dw + 1 < KW9)
                __builtin_prefetch((const char*)fragHi +
                                   (size_t)(((dw + 1) * 4) * 8 + ng * 2) * 1024 + lane * 32, 0, 3);
            #pragma unroll
            for (int cg = 0; cg < 4; ++cg) {
                FragU ahi[2], alo[2];
                #pragma unroll
                for (int mt = 0; mt < 2; ++mt) {
                    const int arow = (mg * 2 + mt) * 16 + li + dw;   // padded row
                    size_t aoff = (size_t)arow * (PITCH * 2) + (size_t)(cg * 32 + half * 8) * 2;
                    ahi[mt].q[0] = *(const uint4*)(Ah + aoff);
                    ahi[mt].q[1] = *(const uint4*)(Ah + aoff + 32);
                    alo[mt].q[0] = *(const uint4*)(Al + aoff);
                    alo[mt].q[1] = *(const uint4*)(Al + aoff + 32);
                }
                const size_t bbase = (size_t)((dw * 4 + cg) * 8 + ng * 2) * 1024 + lane * 32;
                const char* bh = (const char*)fragHi + bbase;
                const char* bl = (const char*)fragLo + bbase;
                FragU bhv[2], blv[2];
                #pragma unroll
                for (int nt = 0; nt < 2; ++nt) {
                    bhv[nt].q[0] = *(const uint4*)(bh + nt * 1024);
                    bhv[nt].q[1] = *(const uint4*)(bh + nt * 1024 + 16);
                    blv[nt].q[0] = *(const uint4*)(bl + nt * 1024);
                    blv[nt].q[1] = *(const uint4*)(bl + nt * 1024 + 16);
                }
                #pragma unroll
                for (int mt = 0; mt < 2; ++mt)
                    #pragma unroll
                    for (int nt = 0; nt < 2; ++nt)
                        acc[mt][nt] = __builtin_amdgcn_wmma_f32_16x16x32_bf16(
                            false, ahi[mt].v, false, bhv[nt].v, (short)0, acc[mt][nt], false, false);
                #pragma unroll
                for (int mt = 0; mt < 2; ++mt)
                    #pragma unroll
                    for (int nt = 0; nt < 2; ++nt)
                        acc[mt][nt] = __builtin_amdgcn_wmma_f32_16x16x32_bf16(
                            false, alo[mt].v, false, bhv[nt].v, (short)0, acc[mt][nt], false, false);
                #pragma unroll
                for (int mt = 0; mt < 2; ++mt)
                    #pragma unroll
                    for (int nt = 0; nt < 2; ++nt)
                        acc[mt][nt] = __builtin_amdgcn_wmma_f32_16x16x32_bf16(
                            false, ahi[mt].v, false, blv[nt].v, (short)0, acc[mt][nt], false, false);
            }
        }

        // epilogue: y[u] = cur + relu(conv); refill next LDS buffer
        __bf16* nhi = (__bf16*)(smem + (size_t)((1 - p) * 2) * PART_BYTES);
        __bf16* nlo = (__bf16*)(smem + (size_t)((1 - p) * 2 + 1) * PART_BYTES);
        #pragma unroll
        for (int mt = 0; mt < 2; ++mt)
            #pragma unroll
            for (int nt = 0; nt < 2; ++nt) {
                const int co = (ng * 2 + nt) * 16 + li;
                #pragma unroll
                for (int r = 0; r < 8; ++r) {
                    const int w = (mg * 2 + mt) * 16 + half * 8 + r;
                    float conv = acc[mt][nt][r];
                    conv = conv > 0.f ? conv : 0.f;
                    float val = cur[mt][nt][r] + conv;
                    Yrow[(size_t)w * sv + co] = val;
                    __bf16 h = (__bf16)val;
                    nhi[(w + 4) * PITCH + co] = h;
                    nlo[(w + 4) * PITCH + co] = (__bf16)(val - (float)h);
                }
            }
        __syncthreads();
        p ^= 1;
    }
}

// ---------------------------------------------------------------------------
extern "C" void kernel_launch(void* const* d_in, const int* in_sizes, int n_in,
                              void* d_out, int out_size, void* d_ws, size_t ws_size,
                              hipStream_t stream)
{
    const float* x  = (const float*)d_in[0];
    const float* kk[4] = {(const float*)d_in[1], (const float*)d_in[2],
                          (const float*)d_in[3], (const float*)d_in[4]};
    float* out = (float*)d_out;

    char* ws = (char*)d_ws;
    float* Y = (float*)ws;                                   // 67,108,864 B NHWC workspace
    const size_t yBytes = (size_t)NB * PLANE * sizeof(float);
    const size_t FRAG_ELEMS = 288 * 512;                     // bf16 elements per blob set
    __bf16* fh[4]; __bf16* fl[4];
    for (int i = 0; i < 4; ++i) {
        fh[i] = (__bf16*)(ws + yBytes) + (size_t)i * 2 * FRAG_ELEMS;
        fl[i] = fh[i] + FRAG_ELEMS;
    }

    for (int i = 0; i < 4; ++i)
        scnn_wprep<<<(288 * 32 + 255) / 256, 256, 0, stream>>>(kk[i], fh[i], fl[i]);

    scnn_t_in<<<dim3(NB * HW, 16), dim3(32, 8), 0, stream>>>(x, Y);

    const int SMEM = 4 * PART_BYTES;                          // 147,968 B of 320 KB/WGP
    scnn_scan<<<NB, 512, SMEM, stream>>>(Y, fh[0], fl[0], 16384, 128, 0);  // top-down
    scnn_scan<<<NB, 512, SMEM, stream>>>(Y, fh[1], fl[1], 16384, 128, 1);  // down-top
    scnn_scan<<<NB, 512, SMEM, stream>>>(Y, fh[2], fl[2], 128, 16384, 0);  // left-right
    scnn_scan<<<NB, 512, SMEM, stream>>>(Y, fh[3], fl[3], 128, 16384, 1);  // right-left

    scnn_t_out<<<dim3(NB * HW, 16), dim3(32, 8), 0, stream>>>(Y, out);
}
